// QuantLinearW4A4Tl_79182017069368
// MI455X (gfx1250) — 8-bit matrix operand order repaired and checked by reading, compiled, not run on hardware
//
#include <hip/hip_runtime.h>

typedef __attribute__((ext_vector_type(8))) int   v8i;
typedef __attribute__((ext_vector_type(8))) float v8f;
typedef __attribute__((ext_vector_type(2))) float v2f;
typedef __attribute__((ext_vector_type(2))) int   v2i;

#define MT 128
#define NT 128
#define KC 128
#define STAGE_DW 4096

template <bool IS_B>
__device__ __forceinline__ void stage_operand(const int* __restrict__ gsrc,
                                              int rowStrideI32,
                                              int rowBase,
                                              int colBaseI32,
                                              unsigned* __restrict__ ldsDst,
                                              int tid)
{
#pragma unroll
    for (int i = 0; i < 16; ++i) {
        const int t    = i * 256 + tid;
        const int v    = t & 7;
        const int L    = (t >> 3) & 31;
        const int frag = t >> 8;
        const int blk16 = frag >> 1;
        const int ks    = frag & 1;
        const int g  = L >> 4;
        const int rl = L & 15;
        const int klocal = IS_B ? (ks * 64 + (v >> 1) * 16 + (v & 1) * 4 + g * 8)
                                : (ks * 64 + (v >> 1) * 16 + (v & 1) * 4 + g * 8);
        const int row = rowBase + blk16 * 16 + rl;
        const int* p = gsrc + (long)row * rowStrideI32 + colBaseI32 + (klocal >> 1);
        const v2i pk = *(const v2i*)p;
        const int b0 = pk[0] & 0xFF;
        const int b1 = pk[1] & 0xFF;
        const int e0 = ((b0 & 0xF) ^ 8) - 8;
        const int e1 = (((b0 >> 4) & 0xF) ^ 8) - 8;
        const int e2 = ((b1 & 0xF) ^ 8) - 8;
        const int e3 = (((b1 >> 4) & 0xF) ^ 8) - 8;
        const unsigned packed = (unsigned)(e0 & 0xFF)
                              | ((unsigned)(e1 & 0xFF) << 8)
                              | ((unsigned)(e2 & 0xFF) << 16)
                              | ((unsigned)(e3 & 0xFF) << 24);
        ldsDst[t] = packed;
    }
}

__global__ __launch_bounds__(256)
void w4a4_lowrank_wmma(const int* __restrict__ xq,
                       const float* __restrict__ xs,
                       const int* __restrict__ wq,
                       const float* __restrict__ wsc,
                       const float* __restrict__ bias,
                       const float* __restrict__ xr,
                       const float* __restrict__ wout,
                       float* __restrict__ out,
                       int M, int N, int K, int R)
{
    __shared__ unsigned ldsA[2][STAGE_DW];
    __shared__ unsigned ldsB[2][STAGE_DW];

    const int tid  = threadIdx.x;
    const int lane = tid & 31;
    const int wave = tid >> 5;
    const int wm   = wave >> 2;
    const int wn   = wave & 3;
    const int g    = lane >> 4;
    const int rl   = lane & 15;

    const int blockN  = blockIdx.x * NT;
    const int blockM  = blockIdx.y * MT;
    const int strideI = K >> 1;
    const int kStages = K / KC;

    v8i acc[4][2];
#pragma unroll
    for (int a = 0; a < 4; ++a)
#pragma unroll
        for (int b = 0; b < 2; ++b)
            acc[a][b] = (v8i)0;

    stage_operand<false>(xq, strideI, blockM, 0, &ldsA[0][0], tid);
    stage_operand<true >(wq, strideI, blockN, 0, &ldsB[0][0], tid);
    __syncthreads();

    for (int kc = 0; kc < kStages; ++kc) {
        const int cur = kc & 1;
        if (kc + 1 < kStages) {
            const int col = (kc + 1) * (KC / 2);
            stage_operand<false>(xq, strideI, blockM, col, &ldsA[cur ^ 1][0], tid);
            stage_operand<true >(wq, strideI, blockN, col, &ldsB[cur ^ 1][0], tid);
        }
#pragma unroll
        for (int ksub = 0; ksub < 2; ++ksub) {
            v8i bfr[2];
#pragma unroll
            for (int ni = 0; ni < 2; ++ni) {
                const int frag = (wn * 2 + ni) * 2 + ksub;
                bfr[ni] = *(const v8i*)&ldsB[cur][frag * 256 + lane * 8];
            }
#pragma unroll
            for (int mi = 0; mi < 4; ++mi) {
                const int frag = (wm * 4 + mi) * 2 + ksub;
                const v8i afr = *(const v8i*)&ldsA[cur][frag * 256 + lane * 8];
#pragma unroll
                for (int ni = 0; ni < 2; ++ni) {
                    acc[mi][ni] = __builtin_amdgcn_wmma_i32_16x16x64_iu8(
                        true, afr, true, bfr[ni], acc[mi][ni], false, false);
                }
            }
        }
        __syncthreads();
    }

    v8f accf[4][2];
#pragma unroll
    for (int mi = 0; mi < 4; ++mi) {
        const int mbase = blockM + (wm * 4 + mi) * 16 + g * 8;
        const v8f xsv = *(const v8f*)(xs + mbase);
#pragma unroll
        for (int ni = 0; ni < 2; ++ni) {
            const int n  = blockN + (wn * 2 + ni) * 16 + rl;
            const float w = wsc[n];
            const float b = bias[n];
#pragma unroll
            for (int j = 0; j < 8; ++j)
                accf[mi][ni][j] = (float)acc[mi][ni][j] * xsv[j] * w + b;
        }
    }

    for (int r = 0; r < R; r += 4) {
        v2f bfr[2];
#pragma unroll
        for (int ni = 0; ni < 2; ++ni) {
            const int n = blockN + (wn * 2 + ni) * 16 + rl;
            bfr[ni] = *(const v2f*)(wout + (long)n * R + r + g * 2);
        }
#pragma unroll
        for (int mi = 0; mi < 4; ++mi) {
            const int m = blockM + (wm * 4 + mi) * 16 + rl;
            const v2f afr = *(const v2f*)(xr + (long)m * R + r + g * 2);
#pragma unroll
            for (int ni = 0; ni < 2; ++ni) {
                accf[mi][ni] = __builtin_amdgcn_wmma_f32_16x16x4_f32(
                    false, afr, false, bfr[ni], (short)0, accf[mi][ni], false, false);
            }
        }
    }

#pragma unroll
    for (int mi = 0; mi < 4; ++mi) {
        const int mbase = blockM + (wm * 4 + mi) * 16 + g * 8;
#pragma unroll
        for (int ni = 0; ni < 2; ++ni) {
            const int n = blockN + (wn * 2 + ni) * 16 + rl;
#pragma unroll
            for (int j = 0; j < 8; ++j)
                out[(long)(mbase + j) * N + n] = accf[mi][ni][j];
        }
    }
}

extern "C" void kernel_launch(void* const* d_in, const int* in_sizes, int n_in,
                              void* d_out, int out_size, void* d_ws, size_t ws_size,
                              hipStream_t stream)
{
    const int*   xq   = (const int*)d_in[0];
    const float* xs   = (const float*)d_in[1];
    const int*   wq   = (const int*)d_in[2];
    const float* wsc  = (const float*)d_in[3];
    const float* bias = (const float*)d_in[4];
    const float* xr   = (const float*)d_in[5];
    const float* wout = (const float*)d_in[6];
    float*       out  = (float*)d_out;

    const int M = in_sizes[1];
    const int N = in_sizes[4];
    const int K = 2 * (in_sizes[0] / M);
    const int R = in_sizes[5] / M;

    dim3 grid(N / NT, M / MT);
    w4a4_lowrank_wmma<<<grid, 256, 0, stream>>>(xq, xs, wq, wsc, bias, xr, wout,
                                                out, M, N, K, R);
}
